// Mamba_25202868093266
// MI455X (gfx1250) — matrix operand order repaired and checked by reading, not compiled, not run on hardware
//
#include <hip/hip_runtime.h>

typedef __attribute__((ext_vector_type(16))) __bf16          v16bf;
typedef __attribute__((ext_vector_type(8)))  float           v8f;
typedef __attribute__((ext_vector_type(8)))  unsigned short  ushort8;

union FragBF { ushort8 u8[2]; v16bf v; };

static __device__ __forceinline__ unsigned short f2bf(float f) {
    unsigned int u = __float_as_uint(f);
    u += 0x7FFFu + ((u >> 16) & 1u);          // round-to-nearest-even
    return (unsigned short)(u >> 16);
}

static __device__ __forceinline__ unsigned lds_off(const void* p) {
    // Generic pointers to LDS carry the wave-relative LDS byte address in
    // their low 32 bits.
    return (unsigned)(unsigned long long)p;
}

static __device__ __forceinline__ void async_ld_b128(unsigned ldsaddr,
                                                     const void* gaddr) {
    asm volatile("global_load_async_to_lds_b128 %0, %1, off"
                 :: "v"(ldsaddr), "v"((unsigned long long)gaddr)
                 : "memory");
}

static __device__ __forceinline__ void wait_async0() {
    asm volatile("s_wait_asynccnt 0x0" ::: "memory");
}

// ---------------------------------------------------------------------------
// WMMA GEMM:  C[M,N] = A[M,K](bf16) * B[N,K](bf16)^T  (+ bias[N]) (+ resid)
// Block tile 128x128, K-tile 32 (== WMMA K). 8 waves, each owns 64x32 (4x2
// v8f accumulators). Double-buffered LDS staged with GLOBAL_LOAD_ASYNC_TO_LDS
// (ASYNCcnt), so the DMA for K-tile i+1 overlaps the WMMAs for K-tile i.
// Per-thread staging addresses are hoisted out of the K-loop (stage address
// is base + 2*k0); the pipeline is peeled so the steady-state loop issues
// unconditionally. OOB rows (N=96 tile) are pre-zeroed in LDS and OOB lanes
// redirect to a dummy LDS slot (buffer step = 0) so EXEC stays full.
// ---------------------------------------------------------------------------
__global__ __launch_bounds__(256) void wmma_gemm_bf16(
    const unsigned short* __restrict__ A, int lda,
    const unsigned short* __restrict__ Bw,            // [N, K], ldb == K
    const float* __restrict__ bias,                   // [N] or nullptr
    const float* __restrict__ resid,                  // [M*ldc] or nullptr
    float* __restrict__ Cm, int ldc,
    int M, int Ncols, int K)
{
    __shared__ __align__(16) unsigned short As[2][128][40];  // 80B rows (16B aligned)
    __shared__ __align__(16) unsigned short Bs[2][128][40];
    __shared__ __align__(16) unsigned short dummy[8];        // OOB async sink

    const int tid  = threadIdx.x;
    const int lane = tid & 31;
    const int wave = tid >> 5;
    const int wm   = wave >> 2;       // 0..1  -> 64-row half
    const int wn   = wave & 3;        // 0..3  -> 32-col quarter
    const int m15  = lane & 15;
    const int kh   = lane >> 4;       // K-half selector

    const int mBase = blockIdx.y * 128;
    const int nBase = blockIdx.x * 128;
    const unsigned dummyOff = lds_off(&dummy[0]);
    const unsigned BUFSZ = 128u * 40u * 2u;                  // 10240 B per buffer

    // ---- pre-zero LDS rows whose global row is out of range (both buffers) ---
    for (int c = tid; c < 512; c += 256) {
        int r = c >> 2, cc = (c & 3) * 8;
        ushort8 z = {};
        if (mBase + r >= M)     { *(ushort8*)&As[0][r][cc] = z; *(ushort8*)&As[1][r][cc] = z; }
        if (nBase + r >= Ncols) { *(ushort8*)&Bs[0][r][cc] = z; *(ushort8*)&Bs[1][r][cc] = z; }
    }

    // ---- hoisted per-thread staging addresses (2 A chunks + 2 B chunks) -----
    unsigned long long gA[2], gB[2];
    unsigned lA[2], lB[2], sA[2], sB[2];
    #pragma unroll
    for (int i = 0; i < 2; ++i) {
        int c = tid + i * 256;
        int r = c >> 2, cc = (c & 3) * 8;
        int gm = mBase + r;
        bool okA = gm < M;
        gA[i] = (unsigned long long)(A + (size_t)(okA ? gm : 0) * lda + cc);
        lA[i] = okA ? lds_off(&As[0][r][cc]) : dummyOff;
        sA[i] = okA ? BUFSZ : 0u;
        int gn = nBase + r;
        bool okB = gn < Ncols;
        gB[i] = (unsigned long long)(Bw + (size_t)(okB ? gn : 0) * K + cc);
        lB[i] = okB ? lds_off(&Bs[0][r][cc]) : dummyOff;
        sB[i] = okB ? BUFSZ : 0u;
    }

    auto issue = [&](int b, int k0) {
        size_t koff = (size_t)k0 * 2;                        // bytes
        #pragma unroll
        for (int i = 0; i < 2; ++i) {
            async_ld_b128(lA[i] + (unsigned)b * sA[i], (const void*)(gA[i] + koff));
            async_ld_b128(lB[i] + (unsigned)b * sB[i], (const void*)(gB[i] + koff));
        }
    };

    v8f acc[4][2] = {};

    auto compute = [&](int b) {
        // A 16x32 bf16: lanes 0-15 rows M=0..15; elems 0-7 -> K = e + 8*kh,
        //               elems 8-15 -> K = 16 + 8*kh + (e-8)
        FragBF a[4];
        #pragma unroll
        for (int mt = 0; mt < 4; ++mt) {
            const unsigned short* rp = &As[b][wm * 64 + mt * 16 + m15][0];
            a[mt].u8[0] = *(const ushort8*)(rp + 8 * kh);
            a[mt].u8[1] = *(const ushort8*)(rp + 16 + 8 * kh);
        }
        // B 32x16 bf16: N = lane&15, K same as A (e + 8*kh; 16 + 8*kh + (e-8))
        FragBF bfr[2];
        #pragma unroll
        for (int nt = 0; nt < 2; ++nt) {
            const unsigned short* rp = &Bs[b][wn * 32 + nt * 16 + m15][0];
            bfr[nt].u8[0] = *(const ushort8*)(rp + 8 * kh);
            bfr[nt].u8[1] = *(const ushort8*)(rp + 16 + 8 * kh);
        }
        #pragma unroll
        for (int mt = 0; mt < 4; ++mt)
            #pragma unroll
            for (int nt = 0; nt < 2; ++nt)
                acc[mt][nt] = __builtin_amdgcn_wmma_f32_16x16x32_bf16(
                    false, a[mt].v, false, bfr[nt].v,
                    (short)0, acc[mt][nt], false, false);
    };

    // ---- software pipeline: DMA(k+1) under WMMA(k); last tile peeled -------
    issue(0, 0);
    int buf = 0;
    int k0 = 0;
    for (; k0 + 32 < K; k0 += 32) {
        wait_async0();          // this wave's stage-k0 loads landed in LDS
        __syncthreads();        // all waves' loads landed; prev compute done
        issue(buf ^ 1, k0 + 32);
        compute(buf);
        buf ^= 1;
    }
    wait_async0();
    __syncthreads();
    compute(buf);

    // ---- epilogue: C layout VGPR r -> M = base + r + 8*kh, N = base + (lane&15)
    #pragma unroll
    for (int mt = 0; mt < 4; ++mt) {
        #pragma unroll
        for (int nt = 0; nt < 2; ++nt) {
            int gn = nBase + wn * 32 + nt * 16 + m15;
            if (gn >= Ncols) continue;
            float bv = bias ? bias[gn] : 0.0f;
            int gm0 = mBase + wm * 64 + mt * 16 + 8 * kh;
            #pragma unroll
            for (int r = 0; r < 8; ++r) {
                int gm = gm0 + r;
                if (gm >= M) continue;
                size_t off = (size_t)gm * ldc + gn;
                float v = acc[mt][nt][r] + bv;
                if (resid) v += resid[off];
                Cm[off] = v;
            }
        }
    }
}

// ---------------------------------------------------------------------------
__global__ void cast_f32_bf16(const float* __restrict__ src,
                              unsigned short* __restrict__ dst, int count)
{
    int i = blockIdx.x * 256 + threadIdx.x;
    if (i < count) dst[i] = f2bf(src[i]);
}

// ---------------------------------------------------------------------------
__global__ __launch_bounds__(256) void rmsnorm_kernel(
    const float* __restrict__ x, const float* __restrict__ w,
    unsigned short* __restrict__ obf, float* __restrict__ of32, int d)
{
    __shared__ float red[256];
    int row = blockIdx.x;
    const float* xr = x + (size_t)row * d;
    float ss = 0.f;
    for (int i = threadIdx.x; i < d; i += 256) { float v = xr[i]; ss += v * v; }
    red[threadIdx.x] = ss;
    __syncthreads();
    for (int s = 128; s > 0; s >>= 1) {
        if (threadIdx.x < s) red[threadIdx.x] += red[threadIdx.x + s];
        __syncthreads();
    }
    float scale = rsqrtf(red[0] / (float)d + 1e-5f);
    for (int i = threadIdx.x; i < d; i += 256) {
        float v = xr[i] * scale * w[i];
        if (obf)  obf[(size_t)row * d + i]  = f2bf(v);
        if (of32) of32[(size_t)row * d + i] = v;
    }
}

// ---------------------------------------------------------------------------
// Depthwise conv1d (k=5, pad=2) over xs = xr[:, :, 0:DI], then SiLU.
__global__ void conv_silu_kernel(
    const float* __restrict__ xr, const float* __restrict__ cw,
    const float* __restrict__ cb, float* __restrict__ xsf,
    unsigned short* __restrict__ xsb, int Ls, int ldr, int DI, int count)
{
    int idx = blockIdx.x * 256 + threadIdx.x;
    if (idx >= count) return;
    int d  = idx % DI;
    int bl = idx / DI;
    int l  = bl % Ls;
    int bb = bl / Ls;
    float acc = cb[d];
    #pragma unroll
    for (int j = 0; j < 5; ++j) {
        int ll = l + j - 2;
        if (ll >= 0 && ll < Ls)
            acc += xr[(size_t)(bb * Ls + ll) * ldr + d] * cw[d * 5 + j];
    }
    float s = acc / (1.f + __expf(-acc));   // SiLU
    xsf[idx] = s;
    xsb[idx] = f2bf(s);
}

// ---------------------------------------------------------------------------
__global__ void softplus_kernel(float* __restrict__ x, int count)
{
    int i = blockIdx.x * 256 + threadIdx.x;
    if (i < count) {
        float v = x[i];
        x[i] = (v > 20.f) ? v : log1pf(__expf(v));
    }
}

// ---------------------------------------------------------------------------
// Selective scan: one thread per (batch, channel); 16 f32 states in registers.
// B_t/C_t (16+16 floats per (b,l)) broadcast through LDS. Fuses +u*D and
// SiLU(res) gating; emits bf16 y for the out_proj WMMA GEMM.
__global__ __launch_bounds__(256) void scan_kernel(
    const float* __restrict__ delta, const float* __restrict__ u,
    const float* __restrict__ xdbl,  const float* __restrict__ xr,
    const float* __restrict__ A_log, const float* __restrict__ Dp,
    unsigned short* __restrict__ ybf,
    int Ls, int DI, int xdld, int bcoff)
{
    int bb = blockIdx.y;
    int d  = blockIdx.x * 256 + threadIdx.x;
    __shared__ float sBC[32];

    float Arow[16];
    #pragma unroll
    for (int j = 0; j < 16; ++j) Arow[j] = -__expf(A_log[d * 16 + j]);
    float hst[16] = {};
    float Dd = Dp[d];

    for (int l = 0; l < Ls; ++l) {
        size_t row = (size_t)bb * Ls + l;
        if (threadIdx.x < 32)
            sBC[threadIdx.x] = xdbl[row * xdld + bcoff + threadIdx.x];
        __syncthreads();

        float dt = delta[row * DI + d];
        float uu = u[row * DI + d];
        float du = dt * uu;
        float y  = 0.f;
        #pragma unroll
        for (int j = 0; j < 16; ++j) {
            float dA = __expf(dt * Arow[j]);
            hst[j] = dA * hst[j] + du * sBC[j];
            y += hst[j] * sBC[16 + j];
        }
        y += uu * Dd;
        float rg = xr[row * (size_t)(2 * DI) + DI + d];   // res half
        y *= rg / (1.f + __expf(-rg));                    // * SiLU(res)
        ybf[row * DI + d] = f2bf(y);
        __syncthreads();
    }
}

// ---------------------------------------------------------------------------
extern "C" void kernel_launch(void* const* d_in, const int* in_sizes, int n_in,
                              void* d_out, int out_size, void* d_ws, size_t ws_size,
                              hipStream_t stream)
{
    (void)in_sizes; (void)n_in; (void)out_size; (void)ws_size;

    const int Bb = 4, Ls = 1024, Dm = 1024, DI = 2048, DTR = 64, NS = 16, LAY = 2;
    const int BL = Bb * Ls;              // 4096 rows
    const int XD = DTR + 2 * NS;         // 96

    const float* x     = (const float*)d_in[0];
    const float* Wi    = (const float*)d_in[1];
    const float* bi    = (const float*)d_in[2];
    const float* cw    = (const float*)d_in[3];
    const float* cb    = (const float*)d_in[4];
    const float* Wx    = (const float*)d_in[5];
    const float* Wdt   = (const float*)d_in[6];
    const float* bdt   = (const float*)d_in[7];
    const float* A_log = (const float*)d_in[8];
    const float* Dp    = (const float*)d_in[9];
    const float* Wo    = (const float*)d_in[10];
    const float* bo    = (const float*)d_in[11];
    const float* nw    = (const float*)d_in[12];
    const float* nfw   = (const float*)d_in[13];

    char* wp = (char*)d_ws;
    auto alloc = [&](size_t bytes) {
        void* p = (void*)wp;
        wp += (bytes + 255) & ~(size_t)255;
        return p;
    };

    float*          hbuf    = (float*)         alloc((size_t)BL * Dm * 4);
    unsigned short* xn_bf   = (unsigned short*)alloc((size_t)BL * Dm * 2);
    unsigned short* Wi_bf   = (unsigned short*)alloc((size_t)2 * DI * Dm * 2);
    float*          xrbuf   = (float*)         alloc((size_t)BL * 2 * DI * 4);
    float*          xs_f    = (float*)         alloc((size_t)BL * DI * 4);
    unsigned short* xs_bf   = (unsigned short*)alloc((size_t)BL * DI * 2);
    unsigned short* Wx_bf   = (unsigned short*)alloc((size_t)XD * DI * 2);
    float*          xdbl    = (float*)         alloc((size_t)BL * XD * 4);
    unsigned short* xdbl_bf = (unsigned short*)alloc((size_t)BL * XD * 2);
    unsigned short* Wdt_bf  = (unsigned short*)alloc((size_t)DI * DTR * 2);
    float*          dbuf    = (float*)         alloc((size_t)BL * DI * 4);
    unsigned short* y_bf    = (unsigned short*)alloc((size_t)BL * DI * 2);
    unsigned short* Wo_bf   = (unsigned short*)alloc((size_t)Dm * DI * 2);

    hipMemcpyAsync(hbuf, x, (size_t)BL * Dm * 4, hipMemcpyDeviceToDevice, stream);

    auto gemm = [&](const unsigned short* Ab, int lda, const unsigned short* Bm,
                    const float* bias, const float* resid, float* Cm, int ldc,
                    int M, int Ncols, int K) {
        dim3 grid((Ncols + 127) / 128, M / 128);
        wmma_gemm_bf16<<<grid, 256, 0, stream>>>(Ab, lda, Bm, bias, resid,
                                                 Cm, ldc, M, Ncols, K);
    };
    auto cast = [&](const float* s, unsigned short* dst, size_t cnt) {
        cast_f32_bf16<<<(unsigned)((cnt + 255) / 256), 256, 0, stream>>>(s, dst, (int)cnt);
    };

    for (int i = 0; i < LAY; ++i) {
        // rmsnorm -> bf16 activations
        rmsnorm_kernel<<<BL, 256, 0, stream>>>(hbuf, nw + (size_t)i * Dm,
                                               xn_bf, nullptr, Dm);
        // in_proj: [4096,1024] x [4096,1024]^T -> [4096,4096]
        cast(Wi + (size_t)i * 2 * DI * Dm, Wi_bf, (size_t)2 * DI * Dm);
        gemm(xn_bf, Dm, Wi_bf, bi + (size_t)i * 2 * DI, nullptr,
             xrbuf, 2 * DI, BL, 2 * DI, Dm);
        // depthwise conv + SiLU on xs half
        conv_silu_kernel<<<(BL * DI) / 256, 256, 0, stream>>>(
            xrbuf, cw + (size_t)i * DI * 5, cb + (size_t)i * DI,
            xs_f, xs_bf, Ls, 2 * DI, DI, BL * DI);
        // x_proj: [4096,2048] x [96,2048]^T -> [4096,96]
        cast(Wx + (size_t)i * XD * DI, Wx_bf, (size_t)XD * DI);
        gemm(xs_bf, DI, Wx_bf, nullptr, nullptr, xdbl, XD, BL, XD, DI);
        // dt_proj on dlt = xdbl[:, :64]: [4096,64] x [2048,64]^T -> [4096,2048]
        cast(xdbl, xdbl_bf, (size_t)BL * XD);
        cast(Wdt + (size_t)i * DI * DTR, Wdt_bf, (size_t)DI * DTR);
        gemm(xdbl_bf, XD, Wdt_bf, bdt + (size_t)i * DI, nullptr,
             dbuf, DI, BL, DI, DTR);
        softplus_kernel<<<(BL * DI) / 256, 256, 0, stream>>>(dbuf, BL * DI);
        // selective scan (fused D*u and SiLU(res) gate), emits bf16 y
        scan_kernel<<<dim3(DI / 256, Bb), 256, 0, stream>>>(
            dbuf, xs_f, xdbl, xrbuf, A_log + (size_t)i * DI * NS,
            Dp + (size_t)i * DI, y_bf, Ls, DI, XD, DTR);
        // out_proj + bias + residual (in-place on hbuf)
        cast(Wo + (size_t)i * Dm * DI, Wo_bf, (size_t)Dm * DI);
        gemm(y_bf, DI, Wo_bf, bo + (size_t)i * Dm, hbuf, hbuf, Dm, BL, Dm, DI);
    }

    // final rmsnorm -> f32 output
    rmsnorm_kernel<<<BL, 256, 0, stream>>>(hbuf, nfw, nullptr, (float*)d_out, Dm);
}
